// Model_80513456931023
// MI455X (gfx1250) — compile-verified, matrix operand order checked by reading, not run on hardware
//
#include <hip/hip_runtime.h>

typedef float    v2f  __attribute__((ext_vector_type(2)));
typedef float    v8f  __attribute__((ext_vector_type(8)));
typedef _Float16 v16h __attribute__((ext_vector_type(16)));

// ---------------------------------------------------------------------------
// K0: zero the agg1 scratch buffer
// ---------------------------------------------------------------------------
__global__ void k_zero(float* __restrict__ p, int n) {
    int i = blockIdx.x * blockDim.x + threadIdx.x;
    if (i < n) p[i] = 0.0f;
}

// ---------------------------------------------------------------------------
// K1: layer-1 message + scatter-add.  agg1[dst] += w_e * x[src]   (width 4)
// ---------------------------------------------------------------------------
__global__ void k_edge_msg1(const float4* __restrict__ x,     // [N] of float4
                            const int*    __restrict__ src,   // [E]
                            const int*    __restrict__ dst,   // [E]
                            const float*  __restrict__ ew,    // [E]
                            float*        __restrict__ agg,   // [N*4]
                            int E) {
    int e = blockIdx.x * blockDim.x + threadIdx.x;
    if (e >= E) return;
    int    s  = src[e];
    int    d  = dst[e];
    float  w  = ew[e];
    float4 xv = x[s];
    float* a  = agg + 4 * d;
    atomicAdd(a + 0, w * xv.x);
    atomicAdd(a + 1, w * xv.y);
    atomicAdd(a + 2, w * xv.z);
    atomicAdd(a + 3, w * xv.w);
}

// ---------------------------------------------------------------------------
// K2: fused node phase via WMMA.
// One wave32 per tile of 16 nodes. For each 16-column slice t of HIDDEN=128:
//   C = b1 (broadcast rows); C += agg1_tile(16x4) @ W1_rel(4x16)   [WMMA #1]
//   C += x_tile(16x4) @ W1_root(4x16)                              [WMMA #2]
//   h = relu(C);  s += h*W2_rel[col];  r += h*W2_root[col]
// Then 16-lane xor reduction -> s[node], out[node] = r + b2.
//
// f32 16x16x4 VGPR layouts (ISA 7.12.2):
//   A:  lane L: M = L%16, K = 2*(L/16) + {vgpr0,vgpr1}
//   B:  lane L: N = L%16, K = 2*(L/16) + {vgpr0,vgpr1}
//   C/D: lane L vgpr v: M = v + 8*(L/16), N = L%16
// ---------------------------------------------------------------------------
__global__ void __launch_bounds__(256)
k_node_wmma(const float* __restrict__ agg,   // [N,4]
            const float* __restrict__ x,     // [N,4]
            const float* __restrict__ W1r,   // [4,128] row-major
            const float* __restrict__ b1,    // [128]
            const float* __restrict__ W1o,   // [4,128]
            const float* __restrict__ W2r,   // [128]
            const float* __restrict__ b2,    // [1]
            const float* __restrict__ W2o,   // [128]
            float*       __restrict__ s_out, // [N]
            float*       __restrict__ out,   // [N]
            int nTiles) {
    const int lane = threadIdx.x & 31;
    const int wid  = threadIdx.x >> 5;
    const int tile = blockIdx.x * (blockDim.x >> 5) + wid;
    if (tile >= nTiles) return;          // wave-uniform: EXEC stays all-1s

    const int base = tile * 16;
    const int r16  = lane & 15;
    const int half = lane >> 4;
    const int k0   = 2 * half;
    const int row  = base + r16;

    const float bias2 = b2[0];

#if __has_builtin(__builtin_amdgcn_wmma_f32_16x16x4_f32)
    // -------- full-precision f32 path: V_WMMA_F32_16X16X4_F32 --------
    v2f a_rel, a_root;
    a_rel.x  = agg[row * 4 + k0];
    a_rel.y  = agg[row * 4 + k0 + 1];
    a_root.x = x[row * 4 + k0];
    a_root.y = x[row * 4 + k0 + 1];
#else
    // -------- fallback: single V_WMMA_F32_16X16X32_F16 per tile,
    // A = [agg1 | x | 0...] (K=8 used), B = [W1_rel ; W1_root ; 0...] --------
    v16h a_cat = {};
    if (half == 0) {
        a_cat[0] = (_Float16)agg[row * 4 + 0];
        a_cat[1] = (_Float16)agg[row * 4 + 1];
        a_cat[2] = (_Float16)agg[row * 4 + 2];
        a_cat[3] = (_Float16)agg[row * 4 + 3];
        a_cat[4] = (_Float16)x[row * 4 + 0];
        a_cat[5] = (_Float16)x[row * 4 + 1];
        a_cat[6] = (_Float16)x[row * 4 + 2];
        a_cat[7] = (_Float16)x[row * 4 + 3];
    }
#endif

    float ps[8] = {0.f, 0.f, 0.f, 0.f, 0.f, 0.f, 0.f, 0.f};
    float pr[8] = {0.f, 0.f, 0.f, 0.f, 0.f, 0.f, 0.f, 0.f};

    for (int t = 0; t < 8; ++t) {
        const int col = t * 16 + r16;
        const float bv = b1[col];
        v8f c = {bv, bv, bv, bv, bv, bv, bv, bv};

#if __has_builtin(__builtin_amdgcn_wmma_f32_16x16x4_f32)
        v2f brel, broot;
        brel.x  = W1r[(k0    ) * 128 + col];
        brel.y  = W1r[(k0 + 1) * 128 + col];
        broot.x = W1o[(k0    ) * 128 + col];
        broot.y = W1o[(k0 + 1) * 128 + col];
        c = __builtin_amdgcn_wmma_f32_16x16x4_f32(
                false, a_rel,  false, brel,  (short)0, c, false, false);
        c = __builtin_amdgcn_wmma_f32_16x16x4_f32(
                false, a_root, false, broot, (short)0, c, false, false);
#else
        v16h b_cat = {};
        if (half == 0) {
            b_cat[0] = (_Float16)W1r[0 * 128 + col];
            b_cat[1] = (_Float16)W1r[1 * 128 + col];
            b_cat[2] = (_Float16)W1r[2 * 128 + col];
            b_cat[3] = (_Float16)W1r[3 * 128 + col];
            b_cat[4] = (_Float16)W1o[0 * 128 + col];
            b_cat[5] = (_Float16)W1o[1 * 128 + col];
            b_cat[6] = (_Float16)W1o[2 * 128 + col];
            b_cat[7] = (_Float16)W1o[3 * 128 + col];
        }
        c = __builtin_amdgcn_wmma_f32_16x16x32_f16(
                false, a_cat, false, b_cat, (short)0, c, false, false);
#endif

        const float w2r = W2r[col];
        const float w2o = W2o[col];
#pragma unroll
        for (int v = 0; v < 8; ++v) {
            float h = c[v] > 0.0f ? c[v] : 0.0f;   // relu
            ps[v] += h * w2r;
            pr[v] += h * w2o;
        }
    }

    // reduce over the 16 lanes of each half-wave (columns within tile)
#pragma unroll
    for (int m = 1; m < 16; m <<= 1) {
#pragma unroll
        for (int v = 0; v < 8; ++v) {
            ps[v] += __shfl_xor(ps[v], m, 16);
            pr[v] += __shfl_xor(pr[v], m, 16);
        }
    }

    if (r16 == 0) {
#pragma unroll
        for (int v = 0; v < 8; ++v) {
            const int node = base + v + 8 * half;   // C-layout row mapping
            s_out[node] = ps[v];
            out[node]   = pr[v] + bias2;
        }
    }
}

// ---------------------------------------------------------------------------
// K3: layer-2 message + scatter-add, pre-folded through W2_rel (scalar/edge).
//     out[dst] += w_e * s[src]
// ---------------------------------------------------------------------------
__global__ void k_edge_msg2(const float* __restrict__ s,
                            const int*   __restrict__ src,
                            const int*   __restrict__ dst,
                            const float* __restrict__ ew,
                            float*       __restrict__ out,
                            int E) {
    int e = blockIdx.x * blockDim.x + threadIdx.x;
    if (e >= E) return;
    atomicAdd(&out[dst[e]], ew[e] * s[src[e]]);
}

// ---------------------------------------------------------------------------
extern "C" void kernel_launch(void* const* d_in, const int* in_sizes, int n_in,
                              void* d_out, int out_size, void* d_ws, size_t ws_size,
                              hipStream_t stream) {
    const float* x   = (const float*)d_in[0];   // [N,4]
    const int*   ei  = (const int*)  d_in[1];   // [2,E]
    const float* ea  = (const float*)d_in[2];   // [E]
    const float* W1r = (const float*)d_in[3];   // [4,128]
    const float* b1  = (const float*)d_in[4];   // [128]
    const float* W1o = (const float*)d_in[5];   // [4,128]
    const float* W2r = (const float*)d_in[6];   // [128]
    const float* b2  = (const float*)d_in[7];   // [1]
    const float* W2o = (const float*)d_in[8];   // [128]
    float* out = (float*)d_out;

    const int N = in_sizes[0] / 4;
    const int E = in_sizes[2];
    const int* src = ei;
    const int* dst = ei + E;

    // workspace layout: agg1 [N*4] floats, then s [N] floats  (~2 MB total)
    float* agg = (float*)d_ws;
    float* s   = agg + (size_t)N * 4;

    // K0: zero agg1
    {
        int n = N * 4;
        k_zero<<<(n + 255) / 256, 256, 0, stream>>>(agg, n);
    }
    // K1: agg1[dst] += w * x[src]
    k_edge_msg1<<<(E + 255) / 256, 256, 0, stream>>>(
        (const float4*)x, src, dst, ea, agg, E);
    // K2: WMMA node phase -> s, out = r + b2
    {
        const int nTiles = (N + 15) / 16;            // 6250 (exact for N=100000)
        const int wavesPerBlock = 8;                 // 256 threads
        const int blocks = (nTiles + wavesPerBlock - 1) / wavesPerBlock;
        k_node_wmma<<<blocks, wavesPerBlock * 32, 0, stream>>>(
            agg, x, W1r, b1, W1o, W2r, b2, W2o, s, out, nTiles);
    }
    // K3: out[dst] += w * s[src]
    k_edge_msg2<<<(E + 255) / 256, 256, 0, stream>>>(s, src, dst, ea, out, E);
}
